// GCBC_4904852652591
// MI455X (gfx1250) — matrix operand order repaired and checked by reading, compiled, not run on hardware
//
#include <hip/hip_runtime.h>
#include <math.h>

// ---------------------------------------------------------------------------
// CDNA5 (gfx1250) transformer forward.
// B=8 N=1024 E=768 H=12 DH=64 DEPTH=6 F=3072 A=7.  M = B*N = 8192.
// All GEMMs run on v_wmma_f32_16x16x32_bf16 (wave32).
// ---------------------------------------------------------------------------

typedef __attribute__((ext_vector_type(16))) __bf16 v16bf;
typedef __attribute__((ext_vector_type(8)))  __bf16 v8bf;
typedef __attribute__((ext_vector_type(8)))  float  v8f;

static constexpr int kB = 8, kN = 1024, kE = 768, kH = 12, kDepth = 6;
static constexpr int kF = 3072, kA = 7, kM = kB * kN;   // 8192

__device__ __forceinline__ unsigned int pack2bf(__bf16 lo, __bf16 hi) {
  const unsigned short a = __builtin_bit_cast(unsigned short, lo);
  const unsigned short b = __builtin_bit_cast(unsigned short, hi);
  return (unsigned int)a | ((unsigned int)b << 16);
}

// ---------------------------------------------------------------------------
// Generic bf16 GEMM:  C[M,N] = epi(A[M,K] @ W[K,N] + bias [, resid])
// Block tile 128x128, BK=64, 256 threads = 8 waves (2x4), wave tile 64x32.
// 16 WMMAs per wave per k-iteration between barrier pairs.
// EPI: 0 = store bf16, 1 = fp32 residual add, 2 = tanh-GELU -> bf16,
//      3 = ReLU -> bf16
// ---------------------------------------------------------------------------
template <int EPI>
__global__ __launch_bounds__(256) void gemm_bf16_kernel(
    const __bf16* __restrict__ A, const __bf16* __restrict__ W,
    const float* __restrict__ bias, const float* __restrict__ resid,
    void* __restrict__ out, int Mdim, int Ndim, int Kdim)
{
  constexpr int LDA = 72;  // bf16 stride: 144B, 16B-aligned, bank-skewed
  __shared__ __align__(16) __bf16 As[128 * LDA];   // [row][k 0..63]
  __shared__ __align__(16) __bf16 Bt[128 * LDA];   // transposed: [col][k 0..63]

  const int tid  = threadIdx.x;
  const int lane = tid & 31;
  const int wave = tid >> 5;
  const int wm   = wave & 1;   // 0..1  (64 rows each)
  const int wn   = wave >> 1;  // 0..3  (32 cols each)
  const int bm   = blockIdx.y * 128;
  const int bn   = blockIdx.x * 128;
  const int l15  = lane & 15;
  const bool hilane = (lane & 16) != 0;

  v8f acc[4][2];
#pragma unroll
  for (int i = 0; i < 4; i++)
#pragma unroll
    for (int j = 0; j < 2; j++) acc[i][j] = {};

  const int arow  = tid >> 1;         // 0..127
  const int ahalf = (tid & 1) * 32;   // k offset 0/32
  const int bkp   = tid >> 3;         // 0..31 -> k pair base 2*bkp
  const int bcg   = (tid & 7) * 16;   // col group base

  for (int k0 = 0; k0 < Kdim; k0 += 64) {
    {  // A tile: 128 rows x 64 k, straight vector copies
      const __bf16* gp = A + (size_t)(bm + arow) * Kdim + k0 + ahalf;
#pragma unroll
      for (int j = 0; j < 4; j++)
        *reinterpret_cast<v8bf*>(&As[arow * LDA + ahalf + j * 8]) =
            *reinterpret_cast<const v8bf*>(gp + j * 8);
    }
    {  // B tile: 64 k x 128 cols; load 2 k-rows, pack (k,k+1) -> b32 stores
      const int kk = 2 * bkp;
      const __bf16* g0 = W + (size_t)(k0 + kk) * Ndim + bn + bcg;
      const __bf16* g1 = g0 + Ndim;
      v8bf r0a = *reinterpret_cast<const v8bf*>(g0);
      v8bf r0b = *reinterpret_cast<const v8bf*>(g0 + 8);
      v8bf r1a = *reinterpret_cast<const v8bf*>(g1);
      v8bf r1b = *reinterpret_cast<const v8bf*>(g1 + 8);
#pragma unroll
      for (int c = 0; c < 8; c++)
        *reinterpret_cast<unsigned int*>(&Bt[(bcg + c) * LDA + kk]) =
            pack2bf(r0a[c], r1a[c]);
#pragma unroll
      for (int c = 0; c < 8; c++)
        *reinterpret_cast<unsigned int*>(&Bt[(bcg + 8 + c) * LDA + kk]) =
            pack2bf(r0b[c], r1b[c]);
    }
    __syncthreads();

#pragma unroll
    for (int kc = 0; kc < 64; kc += 32) {
      // A fragments: lane<16 holds K 0-7 & 16-23; lane>=16 holds 8-15 & 24-31
      v16bf afr[4];
#pragma unroll
      for (int mt = 0; mt < 4; mt++) {
        const int row = wm * 64 + mt * 16 + l15;
        const int ko  = kc + (hilane ? 8 : 0);
        v8bf lo = *reinterpret_cast<const v8bf*>(&As[row * LDA + ko]);
        v8bf hh = *reinterpret_cast<const v8bf*>(&As[row * LDA + ko + 16]);
#pragma unroll
        for (int e = 0; e < 8; e++) { afr[mt][e] = lo[e]; afr[mt][e + 8] = hh[e]; }
      }
      // B fragments: lane<16 holds K 0-7 & 16-23; lane>=16 holds 8-15 & 24-31
      v16bf bfr[2];
#pragma unroll
      for (int nt = 0; nt < 2; nt++) {
        const int col = wn * 32 + nt * 16 + l15;
        const int ko  = kc + (hilane ? 8 : 0);
        v8bf lo = *reinterpret_cast<const v8bf*>(&Bt[col * LDA + ko]);
        v8bf hh = *reinterpret_cast<const v8bf*>(&Bt[col * LDA + ko + 16]);
#pragma unroll
        for (int e = 0; e < 8; e++) { bfr[nt][e] = lo[e]; bfr[nt][e + 8] = hh[e]; }
      }
#pragma unroll
      for (int mt = 0; mt < 4; mt++)
#pragma unroll
        for (int nt = 0; nt < 2; nt++)
          acc[mt][nt] = __builtin_amdgcn_wmma_f32_16x16x32_bf16(
              false, afr[mt], false, bfr[nt], (short)0, acc[mt][nt],
              false, false);
    }
    __syncthreads();
  }

  // Epilogue (C layout: VGPR r -> M = r (+8 for lanes 16-31), N = lane%16)
#pragma unroll
  for (int mt = 0; mt < 4; mt++) {
#pragma unroll
    for (int nt = 0; nt < 2; nt++) {
      const int col = bn + wn * 32 + nt * 16 + l15;
      const float bv = bias ? bias[col] : 0.f;
#pragma unroll
      for (int r = 0; r < 8; r++) {
        const int row = bm + wm * 64 + mt * 16 + r + (hilane ? 8 : 0);
        float v = acc[mt][nt][r] + bv;
        const size_t idx = (size_t)row * Ndim + col;
        if (EPI == 0) {
          ((__bf16*)out)[idx] = (__bf16)v;
        } else if (EPI == 1) {
          ((float*)out)[idx] = resid[idx] + v;
        } else if (EPI == 2) {
          float t = 0.7978845608028654f * (v + 0.044715f * v * v * v);
          ((__bf16*)out)[idx] = (__bf16)(0.5f * v * (1.f + tanhf(t)));
        } else {
          ((__bf16*)out)[idx] = (__bf16)(v > 0.f ? v : 0.f);
        }
      }
    }
  }
}

// ---------------------------------------------------------------------------
// Flash attention with ALiBi + causal mask.
// Grid: (N/64, B*H).  128 threads = 4 waves; each wave owns 16 q rows.
// ---------------------------------------------------------------------------
__device__ __constant__ float c_slopes[12] = {
    0.5f, 0.25f, 0.125f, 0.0625f, 0.03125f, 0.015625f,
    0.0078125f, 0.00390625f,
    0.7071067811865476f, 0.35355339059327373f,
    0.17677669529663687f, 0.08838834764831843f};

__global__ __launch_bounds__(128) void attn_kernel(
    const __bf16* __restrict__ qkv, __bf16* __restrict__ out)
{
  constexpr int LD = 72;  // 144B stride
  __shared__ __align__(16) __bf16 Qs[64 * LD];
  __shared__ __align__(16) __bf16 Ks[64 * LD];   // [key][d]
  __shared__ __align__(16) __bf16 Vt[64 * LD];   // [d][key]
  __shared__ __align__(16) __bf16 Ps[64 * LD];   // per-wave 16-row regions

  const int tid  = threadIdx.x;
  const int lane = tid & 31;
  const int wave = tid >> 5;
  const int l15  = lane & 15;
  const bool hilane = (lane & 16) != 0;
  const int qb = blockIdx.x;
  const int bh = blockIdx.y;
  const int b = bh / kH, h = bh % kH;
  const float slope = c_slopes[h];

  {  // load Q tile 64x64
    const int r  = tid >> 1;
    const int ch = (tid & 1) * 32;
    const __bf16* gp = qkv + (size_t)(b * kN + qb * 64 + r) * (3 * kE) + h * 64 + ch;
#pragma unroll
    for (int j = 0; j < 4; j++)
      *reinterpret_cast<v8bf*>(&Qs[r * LD + ch + j * 8]) =
          *reinterpret_cast<const v8bf*>(gp + j * 8);
  }
  __syncthreads();

  // Hoist Q A-fragments (invariant over key blocks): kc = 0 and 32
  v16bf aq[2];
#pragma unroll
  for (int kci = 0; kci < 2; kci++) {
    const int row = wave * 16 + l15;
    const int ko  = kci * 32 + (hilane ? 8 : 0);
    v8bf lo = *reinterpret_cast<const v8bf*>(&Qs[row * LD + ko]);
    v8bf hh = *reinterpret_cast<const v8bf*>(&Qs[row * LD + ko + 16]);
#pragma unroll
    for (int e = 0; e < 8; e++) { aq[kci][e] = lo[e]; aq[kci][e + 8] = hh[e]; }
  }

  float mrow[8], lrow[8];
#pragma unroll
  for (int r = 0; r < 8; r++) { mrow[r] = -1e30f; lrow[r] = 0.f; }
  v8f oacc[4];
#pragma unroll
  for (int d = 0; d < 4; d++) oacc[d] = {};

  for (int kb = 0; kb <= qb; kb++) {
    {  // load K straight (64 key x 64 d)
      const int r  = tid >> 1;
      const int ch = (tid & 1) * 32;
      const __bf16* gk =
          qkv + (size_t)(b * kN + kb * 64 + r) * (3 * kE) + kE + h * 64 + ch;
#pragma unroll
      for (int j = 0; j < 4; j++)
        *reinterpret_cast<v8bf*>(&Ks[r * LD + ch + j * 8]) =
            *reinterpret_cast<const v8bf*>(gk + j * 8);
    }
    {  // load V transposed: key-pair per thread, packed b32 stores
      const int kk = 2 * (tid >> 2);        // key pair base (0,2,..,62)
      const int dg = (tid & 3) * 16;        // 16 d per thread
      const size_t r0 = (size_t)(b * kN + kb * 64 + kk) * (3 * kE);
      const __bf16* g0 = qkv + r0 + 2 * kE + h * 64 + dg;
      const __bf16* g1 = g0 + 3 * kE;
      v8bf v0a = *reinterpret_cast<const v8bf*>(g0);
      v8bf v0b = *reinterpret_cast<const v8bf*>(g0 + 8);
      v8bf v1a = *reinterpret_cast<const v8bf*>(g1);
      v8bf v1b = *reinterpret_cast<const v8bf*>(g1 + 8);
#pragma unroll
      for (int e = 0; e < 8; e++)
        *reinterpret_cast<unsigned int*>(&Vt[(dg + e) * LD + kk]) =
            pack2bf(v0a[e], v1a[e]);
#pragma unroll
      for (int e = 0; e < 8; e++)
        *reinterpret_cast<unsigned int*>(&Vt[(dg + 8 + e) * LD + kk]) =
            pack2bf(v0b[e], v1b[e]);
    }
    __syncthreads();

    // S = Q K^T : this wave's 16 rows x 64 keys (4 tiles of 16)
    float sv[4][8];
#pragma unroll
    for (int nt = 0; nt < 4; nt++) {
      v8f s = {};
#pragma unroll
      for (int kci = 0; kci < 2; kci++) {
        const int kc = kci * 32;
        v16bf bb;
        const int key = nt * 16 + l15;
        const int ko  = kc + (hilane ? 8 : 0);
        v8bf lo = *reinterpret_cast<const v8bf*>(&Ks[key * LD + ko]);
        v8bf hh = *reinterpret_cast<const v8bf*>(&Ks[key * LD + ko + 16]);
#pragma unroll
        for (int e = 0; e < 8; e++) { bb[e] = lo[e]; bb[e + 8] = hh[e]; }
        s = __builtin_amdgcn_wmma_f32_16x16x32_bf16(false, aq[kci], false, bb,
                                                    (short)0, s, false, false);
      }
      const int key = kb * 64 + nt * 16 + l15;
      const float pb = slope * (float)key;
#pragma unroll
      for (int r = 0; r < 8; r++) {
        const int row = qb * 64 + wave * 16 + r + (hilane ? 8 : 0);
        float v = s[r] * 0.125f + pb;           // dh^-0.5 = 0.125
        sv[nt][r] = (key > row) ? -1e30f : v;   // causal mask
      }
    }

    // Online softmax (row = VGPR idx across a 16-lane half; reduce over lanes)
#pragma unroll
    for (int r = 0; r < 8; r++) {
      float mx = fmaxf(fmaxf(sv[0][r], sv[1][r]), fmaxf(sv[2][r], sv[3][r]));
#pragma unroll
      for (int off = 8; off >= 1; off >>= 1)
        mx = fmaxf(mx, __shfl_xor(mx, off, 16));
      const float mnew = fmaxf(mrow[r], mx);
      const float corr = __expf(mrow[r] - mnew);
      float psum = 0.f;
#pragma unroll
      for (int nt = 0; nt < 4; nt++) {
        const float p = __expf(sv[nt][r] - mnew);
        sv[nt][r] = p;
        psum += p;
      }
#pragma unroll
      for (int off = 8; off >= 1; off >>= 1)
        psum += __shfl_xor(psum, off, 16);
      lrow[r] = lrow[r] * corr + psum;
      mrow[r] = mnew;
#pragma unroll
      for (int d = 0; d < 4; d++) oacc[d][r] *= corr;
    }

    // Stage P (bf16) in this wave's private LDS region (same-wave DS order)
#pragma unroll
    for (int nt = 0; nt < 4; nt++)
#pragma unroll
      for (int r = 0; r < 8; r++)
        Ps[(wave * 16 + r + (hilane ? 8 : 0)) * LD + nt * 16 + l15] =
            (__bf16)sv[nt][r];

    // O += P V
#pragma unroll
    for (int kci = 0; kci < 2; kci++) {
      const int kc = kci * 32;
      v16bf a;
      {
        const int row = wave * 16 + l15;
        const int ko  = kc + (hilane ? 8 : 0);
        v8bf lo = *reinterpret_cast<const v8bf*>(&Ps[row * LD + ko]);
        v8bf hh = *reinterpret_cast<const v8bf*>(&Ps[row * LD + ko + 16]);
#pragma unroll
        for (int e = 0; e < 8; e++) { a[e] = lo[e]; a[e + 8] = hh[e]; }
      }
#pragma unroll
      for (int dt = 0; dt < 4; dt++) {
        v16bf bb;
        const int d  = dt * 16 + l15;
        const int ko = kc + (hilane ? 8 : 0);
        v8bf lo = *reinterpret_cast<const v8bf*>(&Vt[d * LD + ko]);
        v8bf hh = *reinterpret_cast<const v8bf*>(&Vt[d * LD + ko + 16]);
#pragma unroll
        for (int e = 0; e < 8; e++) { bb[e] = lo[e]; bb[e + 8] = hh[e]; }
        oacc[dt] = __builtin_amdgcn_wmma_f32_16x16x32_bf16(
            false, a, false, bb, (short)0, oacc[dt], false, false);
      }
    }
    __syncthreads();
  }

  // Write O / l
#pragma unroll
  for (int dt = 0; dt < 4; dt++)
#pragma unroll
    for (int r = 0; r < 8; r++) {
      const int row = qb * 64 + wave * 16 + r + (hilane ? 8 : 0);
      const int d   = dt * 16 + l15;
      out[(size_t)(b * kN + row) * kE + h * 64 + d] =
          (__bf16)(oacc[dt][r] / lrow[r]);
    }
}

// ---------------------------------------------------------------------------
// LayerNorm: one 256-thread block per row of 768 fp32 -> bf16
// ---------------------------------------------------------------------------
__global__ __launch_bounds__(256) void ln_kernel(
    const float* __restrict__ x, const float* __restrict__ scale,
    const float* __restrict__ bias, __bf16* __restrict__ y)
{
  __shared__ float red[256];
  const int row = blockIdx.x;
  const int tid = threadIdx.x;
  const float* xr = x + (size_t)row * kE;
  const float v0 = xr[tid], v1 = xr[tid + 256], v2 = xr[tid + 512];
  red[tid] = v0 + v1 + v2;
  __syncthreads();
  for (int st = 128; st > 0; st >>= 1) {
    if (tid < st) red[tid] += red[tid + st];
    __syncthreads();
  }
  const float mean = red[0] / (float)kE;
  __syncthreads();
  red[tid] = (v0 - mean) * (v0 - mean) + (v1 - mean) * (v1 - mean) +
             (v2 - mean) * (v2 - mean);
  __syncthreads();
  for (int st = 128; st > 0; st >>= 1) {
    if (tid < st) red[tid] += red[tid + st];
    __syncthreads();
  }
  const float rstd = rsqrtf(red[0] / (float)kE + 1e-6f);
  __bf16* yr = y + (size_t)row * kE;
  yr[tid]       = (__bf16)((v0 - mean) * rstd * scale[tid]       + bias[tid]);
  yr[tid + 256] = (__bf16)((v1 - mean) * rstd * scale[tid + 256] + bias[tid + 256]);
  yr[tid + 512] = (__bf16)((v2 - mean) * rstd * scale[tid + 512] + bias[tid + 512]);
}

// ---------------------------------------------------------------------------
__global__ void f32_to_bf16_kernel(const float* __restrict__ in,
                                   __bf16* __restrict__ out, int n)
{
  const int i = blockIdx.x * 256 + threadIdx.x;
  if (i < n) out[i] = (__bf16)in[i];
}

__global__ void copy_f32_kernel(const float* __restrict__ in,
                                float* __restrict__ out, int n)
{
  const int i = blockIdx.x * 256 + threadIdx.x;
  if (i < n) out[i] = in[i];
}

// Tiny head: out[M,7] = h1[M,768] @ w2[768,7]; one wave per row.
__global__ __launch_bounds__(32) void head2_kernel(
    const __bf16* __restrict__ h1, const float* __restrict__ w2,
    float* __restrict__ out)
{
  const int row  = blockIdx.x;
  const int lane = threadIdx.x;
  float acc[kA] = {};
  const __bf16* hr = h1 + (size_t)row * kE;
  for (int j = lane; j < kE; j += 32) {
    const float v = (float)hr[j];
#pragma unroll
    for (int a = 0; a < kA; a++) acc[a] += v * w2[j * kA + a];
  }
#pragma unroll
  for (int a = 0; a < kA; a++)
#pragma unroll
    for (int off = 16; off >= 1; off >>= 1)
      acc[a] += __shfl_xor(acc[a], off, 32);
  if (lane == 0)
#pragma unroll
    for (int a = 0; a < kA; a++) out[(size_t)row * kA + a] = acc[a];
}

// ---------------------------------------------------------------------------
extern "C" void kernel_launch(void* const* d_in, const int* in_sizes, int n_in,
                              void* d_out, int out_size, void* d_ws,
                              size_t ws_size, hipStream_t stream)
{
  (void)in_sizes; (void)n_in; (void)out_size; (void)ws_size;
  const float* x_in  = (const float*)d_in[0];
  const float* ln1_s = (const float*)d_in[1];
  const float* ln1_b = (const float*)d_in[2];
  const float* wqkv  = (const float*)d_in[3];
  const float* bqkv  = (const float*)d_in[4];
  const float* wo    = (const float*)d_in[5];
  const float* bo    = (const float*)d_in[6];
  const float* ln2_s = (const float*)d_in[7];
  const float* ln2_b = (const float*)d_in[8];
  const float* w1    = (const float*)d_in[9];
  const float* w2    = (const float*)d_in[10];
  const float* lnf_s = (const float*)d_in[11];
  const float* lnf_b = (const float*)d_in[12];
  const float* hw1   = (const float*)d_in[13];
  const float* hb1   = (const float*)d_in[14];
  const float* hw2   = (const float*)d_in[15];

  char* ws = (char*)d_ws;
  size_t off = 0;
  auto alloc = [&](size_t bytes) -> void* {
    void* p = ws + off;
    off += (bytes + 255) & ~(size_t)255;
    return p;
  };
  __bf16* wqkv_bf = (__bf16*)alloc((size_t)kDepth * kE * 3 * kE * 2);
  __bf16* wo_bf   = (__bf16*)alloc((size_t)kDepth * kE * kE * 2);
  __bf16* w1_bf   = (__bf16*)alloc((size_t)kDepth * kE * kF * 2);
  __bf16* w2_bf   = (__bf16*)alloc((size_t)kDepth * kF * kE * 2);
  __bf16* hw1_bf  = (__bf16*)alloc((size_t)kE * kE * 2);
  float*  xbuf    = (float*) alloc((size_t)kM * kE * 4);
  __bf16* ybuf    = (__bf16*)alloc((size_t)kM * kE * 2);
  __bf16* qkvbuf  = (__bf16*)alloc((size_t)kM * 3 * kE * 2);
  __bf16* attnbuf = (__bf16*)alloc((size_t)kM * kE * 2);
  __bf16* hbuf    = (__bf16*)alloc((size_t)kM * kF * 2);

  auto conv = [&](const float* src, __bf16* dst, int n) {
    f32_to_bf16_kernel<<<(n + 255) / 256, 256, 0, stream>>>(src, dst, n);
  };
  conv(wqkv, wqkv_bf, kDepth * kE * 3 * kE);
  conv(wo,   wo_bf,   kDepth * kE * kE);
  conv(w1,   w1_bf,   kDepth * kE * kF);
  conv(w2,   w2_bf,   kDepth * kF * kE);
  conv(hw1,  hw1_bf,  kE * kE);
  copy_f32_kernel<<<(kM * kE + 255) / 256, 256, 0, stream>>>(x_in, xbuf, kM * kE);

  const dim3 blk256(256);
  for (int i = 0; i < kDepth; i++) {
    ln_kernel<<<kM, 256, 0, stream>>>(xbuf, ln1_s + i * kE, ln1_b + i * kE, ybuf);
    gemm_bf16_kernel<0><<<dim3(3 * kE / 128, kM / 128), blk256, 0, stream>>>(
        ybuf, wqkv_bf + (size_t)i * kE * 3 * kE, bqkv + i * 3 * kE, nullptr,
        qkvbuf, kM, 3 * kE, kE);
    attn_kernel<<<dim3(kN / 64, kB * kH), 128, 0, stream>>>(qkvbuf, attnbuf);
    gemm_bf16_kernel<1><<<dim3(kE / 128, kM / 128), blk256, 0, stream>>>(
        attnbuf, wo_bf + (size_t)i * kE * kE, bo + i * kE, xbuf, xbuf,
        kM, kE, kE);
    ln_kernel<<<kM, 256, 0, stream>>>(xbuf, ln2_s + i * kE, ln2_b + i * kE, ybuf);
    gemm_bf16_kernel<2><<<dim3(kF / 128, kM / 128), blk256, 0, stream>>>(
        ybuf, w1_bf + (size_t)i * kE * kF, nullptr, nullptr, hbuf, kM, kF, kE);
    gemm_bf16_kernel<1><<<dim3(kE / 128, kM / 128), blk256, 0, stream>>>(
        hbuf, w2_bf + (size_t)i * kF * kE, nullptr, xbuf, xbuf, kM, kE, kF);
  }
  ln_kernel<<<kM, 256, 0, stream>>>(xbuf, lnf_s, lnf_b, ybuf);
  gemm_bf16_kernel<3><<<dim3(kE / 128, kM / 128), blk256, 0, stream>>>(
      ybuf, hw1_bf, hb1, nullptr, attnbuf, kM, kE, kE);
  head2_kernel<<<kM, 32, 0, stream>>>(attnbuf, hw2, (float*)d_out);
}
